// AdaptiveTopKMoE_76536317215205
// MI455X (gfx1250) — matrix operand order repaired and checked by reading, not compiled, not run on hardware
//
#include <hip/hip_runtime.h>
#include <hip/hip_bf16.h>
#include <math.h>

// ---------------- problem constants (from reference) ----------------
#define TN   8192          // tokens N
#define HD   2048          // hidden H
#define HD2  4096          // 2H
#define NE   8             // experts
#define TOPK 2
#define CAP  1280          // ceil(1.25 * 8192 / 8)
#define NSLOT (NE*CAP)     // 10240 expert slots
#define NDISP (TN*TOPK)    // 16384 dispatches

typedef __attribute__((ext_vector_type(16))) __bf16 v16bf;
typedef __attribute__((ext_vector_type(8)))  float  v8f;

__device__ __forceinline__ unsigned short f2bf(float f) {
  union { float f; unsigned u; } c; c.f = f;
  unsigned u = c.u;
  u += 0x7FFFu + ((u >> 16) & 1u);   // round-to-nearest-even
  return (unsigned short)(u >> 16);
}

// ---------------- 1) router: logits, top-2, softmax ----------------
__global__ __launch_bounds__(256) void router_kernel(
    const float* __restrict__ tokens, const float* __restrict__ Wr,
    const float* __restrict__ br, int* __restrict__ ids, float* __restrict__ pw)
{
  const int wid = threadIdx.x >> 5, lane = threadIdx.x & 31;
  const int t = blockIdx.x * 8 + wid;
  float acc[NE];
  #pragma unroll
  for (int e = 0; e < NE; ++e) acc[e] = 0.f;
  const float* xr = tokens + (size_t)t * HD;
  for (int h = lane; h < HD; h += 32) {
    float x = xr[h];
    #pragma unroll
    for (int e = 0; e < NE; ++e) acc[e] += x * Wr[e * HD + h];
  }
  #pragma unroll
  for (int e = 0; e < NE; ++e) {
    #pragma unroll
    for (int off = 16; off >= 1; off >>= 1)
      acc[e] += __shfl_xor(acc[e], off, 32);
  }
  if (lane == 0) {
    float l[NE];
    #pragma unroll
    for (int e = 0; e < NE; ++e) l[e] = acc[e] + br[e];
    int b0 = 0;
    for (int e = 1; e < NE; ++e) if (l[e] > l[b0]) b0 = e;        // ties: lowest idx
    int b1 = (b0 == 0) ? 1 : 0;
    for (int e = 0; e < NE; ++e) if (e != b0 && l[e] > l[b1]) b1 = e;
    float s0 = l[b0], s1 = l[b1];
    float e1 = expf(s1 - s0);
    float p0 = 1.f / (1.f + e1);
    ids[t * 2] = b0; ids[t * 2 + 1] = b1;
    pw[t * 2] = p0;  pw[t * 2 + 1] = e1 * p0;
  }
}

// ---------------- 2) dispatch: stable per-expert rank + capacity ----------------
__global__ __launch_bounds__(1024) void dispatch_kernel(
    const int* __restrict__ ids, int* __restrict__ slot_token,
    int* __restrict__ token_slot)
{
  __shared__ int sc[1024][NE];      // 32 KB
  const int t = threadIdx.x;
  for (int i = t; i < NSLOT; i += 1024) slot_token[i] = -1;

  unsigned char le[16];
  int cnt[NE];
  #pragma unroll
  for (int e = 0; e < NE; ++e) cnt[e] = 0;
  #pragma unroll
  for (int j = 0; j < 16; ++j) {
    int e = ids[t * 16 + j]; le[j] = (unsigned char)e; cnt[e]++;
  }
  #pragma unroll
  for (int e = 0; e < NE; ++e) sc[t][e] = cnt[e];
  __syncthreads();
  // Hillis-Steele inclusive scan over 1024 threads x 8 counters
  for (int off = 1; off < 1024; off <<= 1) {
    int v[NE];
    if (t >= off) {
      #pragma unroll
      for (int e = 0; e < NE; ++e) v[e] = sc[t - off][e];
    }
    __syncthreads();
    if (t >= off) {
      #pragma unroll
      for (int e = 0; e < NE; ++e) sc[t][e] += v[e];
    }
    __syncthreads();
  }
  int run[NE];
  #pragma unroll
  for (int e = 0; e < NE; ++e) run[e] = t ? sc[t - 1][e] : 0;   // exclusive prefix
  #pragma unroll
  for (int j = 0; j < 16; ++j) {
    int d = t * 16 + j;
    int e = le[j];
    int r = run[e]++;
    int s = -1;
    if (r < CAP) { s = e * CAP + r; slot_token[s] = d >> 1; }   // token = d / K
    token_slot[d] = s;
  }
}

// ---------------- 3) gather tokens -> bf16 Xb ----------------
__global__ __launch_bounds__(256) void gather_kernel(
    const float* __restrict__ tokens, const int* __restrict__ slot_token,
    unsigned short* __restrict__ Xb)
{
  const int s = blockIdx.x;
  const int tk = slot_token[s];
  const int h0 = threadIdx.x * 8;
  union { unsigned short h[8]; uint4 u; } o;
  if (tk < 0) {
    o.u = make_uint4(0u, 0u, 0u, 0u);
  } else {
    const float* xr = tokens + (size_t)tk * HD + h0;
    #pragma unroll
    for (int j = 0; j < 8; ++j) o.h[j] = f2bf(xr[j]);
  }
  *(uint4*)&Xb[(size_t)s * HD + h0] = o.u;
}

// ---------------- 4/5) weight convert+transpose: f32 [K][N] -> bf16 [N][K] ----------------
__global__ __launch_bounds__(256) void wconv_kernel(
    const float* __restrict__ W, unsigned short* __restrict__ Bt, int Kd, int Nd)
{
  __shared__ float tile[64][65];
  const int e = blockIdx.z;
  const int n0 = blockIdx.x * 64, k0 = blockIdx.y * 64;
  const int t = threadIdx.x;
  const float* Wb = W + (size_t)e * Kd * Nd;
  #pragma unroll
  for (int i = 0; i < 4; ++i) {
    int r = (t >> 4) + i * 16;
    int c = (t & 15) * 4;
    const float4 v = *(const float4*)&Wb[(size_t)(k0 + r) * Nd + n0 + c];
    tile[r][c] = v.x; tile[r][c + 1] = v.y; tile[r][c + 2] = v.z; tile[r][c + 3] = v.w;
  }
  __syncthreads();
  unsigned short* Bb = Bt + (size_t)e * Nd * Kd;
  #pragma unroll
  for (int j = 0; j < 2; ++j) {
    int n = t >> 2;
    int kk = (t & 3) * 8 + j * 32;
    union { unsigned short h[8]; uint4 u; } o;
    #pragma unroll
    for (int x = 0; x < 8; ++x) o.h[x] = f2bf(tile[kk + x][n]);
    *(uint4*)&Bb[(size_t)(n0 + n) * Kd + k0 + kk] = o.u;
  }
}

// ---------------- 6/7) WMMA bf16 GEMM: C[M,N] = A[M,K] * Bt[N,K]^T + bias ----------------
// Block tile 256x128, 8 waves in 4(M) x 2(N) grid -> 64x64 per wave (16 WMMA tiles).
#define BM 256
#define BN 128
#define BK 32
#define LP 40   // padded LDS row length in bf16 elems (80B): conflict-free b128 reads

#if defined(__gfx1250__) && __has_builtin(__builtin_amdgcn_global_load_async_to_lds_b128)
#define USE_ASYNC 1
// exact parameter types from clang diagnostic:
//   arg0: int __vector(4) __device__ *   (addrspace(1))
//   arg1: int __vector(4) __shared__ *   (addrspace(3))
typedef int v4i __attribute__((vector_size(4 * sizeof(int))));
typedef __attribute__((address_space(1))) v4i gv4i;
typedef __attribute__((address_space(3))) v4i lv4i;
__device__ __forceinline__ void wait_asynccnt0() {
#if __has_builtin(__builtin_amdgcn_s_wait_asynccnt)
  __builtin_amdgcn_s_wait_asynccnt(0);
#else
  asm volatile("s_wait_asynccnt 0x0" ::: "memory");
#endif
}
#else
#define USE_ASYNC 0
#endif

union Frag { uint4 u[2]; v16bf v; };

template <int ACT>
__global__ __launch_bounds__(256) void moe_gemm_kernel(
    const unsigned short* __restrict__ A,   // bf16 [E][M][Kd]
    const unsigned short* __restrict__ Bt,  // bf16 [E][Nd][Kd]
    const float* __restrict__ bias,         // [E][Nd]
    void* __restrict__ Out,                 // ACT: bf16 [E][M][Nd] ; else f32
    int M, int Nd, int Kd)
{
  const int e = blockIdx.z;
  const int n0 = blockIdx.x * BN, m0 = blockIdx.y * BM;
  const int tid = threadIdx.x, lane = tid & 31, wid = tid >> 5;
  const int wm = wid >> 1, wn = wid & 1;      // 4 x 2 wave grid -> 64x64 per wave
  const int lrow = lane & 15, hi = lane >> 4;
  const int ak0 = hi ? 8 : 0;                 // ISA A-frag K layout per half-wave
  const unsigned short* Ab = A + ((size_t)e * M + m0) * Kd;
  const unsigned short* Bb = Bt + ((size_t)e * Nd + n0) * Kd;

  v8f acc[4][4];
  #pragma unroll
  for (int i = 0; i < 4; ++i)
    #pragma unroll
    for (int j = 0; j < 4; ++j)
      acc[i][j] = (v8f){0.f, 0.f, 0.f, 0.f, 0.f, 0.f, 0.f, 0.f};

#if USE_ASYNC
  // ---- CDNA5 async global->LDS pipeline: no VGPR staging, ASYNCcnt tracked ----
  __shared__ __align__(16) unsigned short As[2][BM * LP];  // 2 x 20 KB
  __shared__ __align__(16) unsigned short Bs[2][BN * LP];  // 2 x 10 KB

  // prologue: async-stage K-tile 0 into buffer 0
  #pragma unroll
  for (int i = 0; i < 4; ++i) {
    int c = tid + i * 256, r = c >> 2, q = (c & 3) * 8;
    __builtin_amdgcn_global_load_async_to_lds_b128(
        (gv4i*)(Ab + (size_t)r * Kd + q), (lv4i*)&As[0][r * LP + q], 0, 0);
  }
  #pragma unroll
  for (int i = 0; i < 2; ++i) {
    int c = tid + i * 256, r = c >> 2, q = (c & 3) * 8;
    __builtin_amdgcn_global_load_async_to_lds_b128(
        (gv4i*)(Bb + (size_t)r * Kd + q), (lv4i*)&Bs[0][r * LP + q], 0, 0);
  }
  wait_asynccnt0();
  __syncthreads();

  const int nK = Kd / BK;
  for (int kt = 0; kt < nK; ++kt) {
    const int cur = kt & 1, nxt = cur ^ 1;
    if (kt + 1 < nK) {                       // async-prefetch next tile (other buffer)
      const int ko = (kt + 1) * BK;
      #pragma unroll
      for (int i = 0; i < 4; ++i) {
        int c = tid + i * 256, r = c >> 2, q = (c & 3) * 8;
        __builtin_amdgcn_global_load_async_to_lds_b128(
            (gv4i*)(Ab + (size_t)r * Kd + ko + q), (lv4i*)&As[nxt][r * LP + q], 0, 0);
      }
      #pragma unroll
      for (int i = 0; i < 2; ++i) {
        int c = tid + i * 256, r = c >> 2, q = (c & 3) * 8;
        __builtin_amdgcn_global_load_async_to_lds_b128(
            (gv4i*)(Bb + (size_t)r * Kd + ko + q), (lv4i*)&Bs[nxt][r * LP + q], 0, 0);
      }
    }
    // compute from current buffer: keep 4 B frags live, stream A frags
    Frag b[4];
    #pragma unroll
    for (int j = 0; j < 4; ++j) {
      int r = (wn * 64 + j * 16 + lrow) * LP;
      b[j].u[0] = *(const uint4*)&Bs[cur][r + hi * 8];
      b[j].u[1] = *(const uint4*)&Bs[cur][r + hi * 8 + 16];
    }
    #pragma unroll
    for (int i = 0; i < 4; ++i) {
      Frag a;
      int r = (wm * 64 + i * 16 + lrow) * LP;
      a.u[0] = *(const uint4*)&As[cur][r + ak0];
      a.u[1] = *(const uint4*)&As[cur][r + ak0 + 16];
      #pragma unroll
      for (int j = 0; j < 4; ++j)
        acc[i][j] = __builtin_amdgcn_wmma_f32_16x16x32_bf16(
            false, a.v, false, b[j].v, (short)0, acc[i][j], false, false);
    }
    wait_asynccnt0();
    __syncthreads();
  }
#else
  // ---- fallback: single-buffer LDS, compiler-pipelined ----
  __shared__ __align__(16) unsigned short As[BM * LP];
  __shared__ __align__(16) unsigned short Bs[BN * LP];

  for (int kt = 0; kt < Kd; kt += BK) {
    #pragma unroll
    for (int i = 0; i < 4; ++i) {
      int c = tid + i * 256, r = c >> 2, q = (c & 3) * 8;
      *(uint4*)&As[r * LP + q] = *(const uint4*)&Ab[(size_t)r * Kd + kt + q];
    }
    #pragma unroll
    for (int i = 0; i < 2; ++i) {
      int c = tid + i * 256, r = c >> 2, q = (c & 3) * 8;
      *(uint4*)&Bs[r * LP + q] = *(const uint4*)&Bb[(size_t)r * Kd + kt + q];
    }
    __syncthreads();

    Frag b[4];
    #pragma unroll
    for (int j = 0; j < 4; ++j) {
      int r = (wn * 64 + j * 16 + lrow) * LP;
      b[j].u[0] = *(const uint4*)&Bs[r + hi * 8];
      b[j].u[1] = *(const uint4*)&Bs[r + hi * 8 + 16];
    }
    #pragma unroll
    for (int i = 0; i < 4; ++i) {
      Frag a;
      int r = (wm * 64 + i * 16 + lrow) * LP;
      a.u[0] = *(const uint4*)&As[r + ak0];
      a.u[1] = *(const uint4*)&As[r + ak0 + 16];
      #pragma unroll
      for (int j = 0; j < 4; ++j)
        acc[i][j] = __builtin_amdgcn_wmma_f32_16x16x32_bf16(
            false, a.v, false, b[j].v, (short)0, acc[i][j], false, false);
    }
    __syncthreads();
  }
#endif

  // epilogue: C/D layout -> elem r8 is row (hi*8 + r8), col lrow of the 16x16 tile
  #pragma unroll
  for (int i = 0; i < 4; ++i) {
    #pragma unroll
    for (int j = 0; j < 4; ++j) {
      int nc = n0 + wn * 64 + j * 16 + lrow;
      float bv = bias[e * Nd + nc];
      #pragma unroll
      for (int r8 = 0; r8 < 8; ++r8) {
        int mr = m0 + wm * 64 + i * 16 + hi * 8 + r8;
        float v = acc[i][j][r8] + bv;
        size_t o = ((size_t)e * M + mr) * Nd + nc;
        if (ACT) {  // exact GELU
          v = 0.5f * v * (1.0f + erff(v * 0.70710678118654752440f));
          ((unsigned short*)Out)[o] = f2bf(v);
        } else {
          ((float*)Out)[o] = v;
        }
      }
    }
  }
}

// ---------------- 8) combine: out[t] = sum_k w_k * Y[slot_k] ----------------
__global__ __launch_bounds__(256) void combine_kernel(
    const int* __restrict__ token_slot, const float* __restrict__ pw,
    const float* __restrict__ Y, float* __restrict__ out)
{
  const int t = blockIdx.x;
  const int s0 = token_slot[t * 2], s1 = token_slot[t * 2 + 1];
  const float w0 = pw[t * 2], w1 = pw[t * 2 + 1];
  const int h0 = threadIdx.x * 8;
  float r[8];
  #pragma unroll
  for (int j = 0; j < 8; ++j) r[j] = 0.f;
  if (s0 >= 0) {
    const float* y = Y + (size_t)s0 * HD + h0;
    #pragma unroll
    for (int j = 0; j < 8; ++j) r[j] += w0 * y[j];
  }
  if (s1 >= 0) {
    const float* y = Y + (size_t)s1 * HD + h0;
    #pragma unroll
    for (int j = 0; j < 8; ++j) r[j] += w1 * y[j];
  }
  float* o = out + (size_t)t * HD + h0;
  #pragma unroll
  for (int j = 0; j < 8; ++j) o[j] = r[j];
}

// ---------------- host launcher ----------------
extern "C" void kernel_launch(void* const* d_in, const int* in_sizes, int n_in,
                              void* d_out, int out_size, void* d_ws, size_t ws_size,
                              hipStream_t stream)
{
  const float* tokens = (const float*)d_in[0];
  const float* Wr     = (const float*)d_in[1];
  const float* br     = (const float*)d_in[2];
  const float* W1     = (const float*)d_in[3];
  const float* b1     = (const float*)d_in[4];
  const float* W2     = (const float*)d_in[5];
  const float* b2     = (const float*)d_in[6];
  float* out = (float*)d_out;
  char* ws = (char*)d_ws;

  // workspace layout (256B aligned offsets), total 478,388,224 bytes (~456 MiB)
  int*            ids        = (int*)(ws + 0);                       //  64 KB
  float*          pw         = (float*)(ws + 65536);                 //  64 KB
  int*            slot_token = (int*)(ws + 131072);                  //  40 KB
  int*            token_slot = (int*)(ws + 172032);                  //  64 KB
  unsigned short* Xb         = (unsigned short*)(ws + 237568);       //  40 MB bf16 [10240][2048]
  unsigned short* Hb         = (unsigned short*)(ws + 42180608);     //  80 MB bf16 [10240][4096]
  float*          Y          = (float*)(ws + 126066688);             //  80 MB f32  [10240][2048]
  unsigned short* Bt1        = (unsigned short*)(ws + 209952768);    // 128 MB bf16 [8][4096][2048]
  unsigned short* Bt2        = (unsigned short*)(ws + 344170496);    // 128 MB bf16 [8][2048][4096]

  router_kernel<<<TN / 8, 256, 0, stream>>>(tokens, Wr, br, ids, pw);
  dispatch_kernel<<<1, 1024, 0, stream>>>(ids, slot_token, token_slot);
  gather_kernel<<<NSLOT, 256, 0, stream>>>(tokens, slot_token, Xb);
  wconv_kernel<<<dim3(HD2 / 64, HD / 64, NE), 256, 0, stream>>>(W1, Bt1, HD, HD2);
  wconv_kernel<<<dim3(HD / 64, HD2 / 64, NE), 256, 0, stream>>>(W2, Bt2, HD2, HD);
  moe_gemm_kernel<1><<<dim3(HD2 / BN, CAP / BM, NE), 256, 0, stream>>>(
      Xb, Bt1, b1, (void*)Hb, CAP, HD2, HD);
  moe_gemm_kernel<0><<<dim3(HD / BN, CAP / BM, NE), 256, 0, stream>>>(
      Hb, Bt2, b2, (void*)Y, CAP, HD, HD2);
  combine_kernel<<<TN, 256, 0, stream>>>(token_slot, pw, Y, out);
}
